// SwinBlock_69415261438540
// MI455X (gfx1250) — matrix operand order repaired and checked by reading, not compiled, not run on hardware
//
#include <hip/hip_runtime.h>
#include <hip/hip_bf16.h>

typedef __attribute__((ext_vector_type(16))) _Float16 v16h;
typedef __attribute__((ext_vector_type(8)))  _Float16 h8;
typedef __attribute__((ext_vector_type(8)))  float    v8f;

union HF16 { v16h v; h8 h2[2]; };

#define BM 64
#define BN 128
#define BK 32

// ------------------------------------------------- CDNA5 async global->LDS
// Per-lane 16B DMA copy, tracked by ASYNCcnt (ISA 15.18.3 op 98).
__device__ __forceinline__ void async_copy_b128(void* lds_ptr, const void* gptr) {
    unsigned lds_off = (unsigned)(uintptr_t)lds_ptr;   // LDS aperture: addr[31:0] = LDS offset
    asm volatile("global_load_async_to_lds_b128 %0, %1, off"
                 :: "v"(lds_off), "v"((unsigned long long)(uintptr_t)gptr)
                 : "memory");
}
__device__ __forceinline__ void wait_async0() {
    asm volatile("s_wait_asynccnt 0x0" ::: "memory");
}

// ---------------------------------------------------------------- utilities
// Convert f32 [K,N] weight to f16 transposed [N,K] (fragment/async friendly).
__global__ void transpose_f32_to_f16(const float* __restrict__ s, _Float16* __restrict__ d,
                                     int K, int N) {
    int i = blockIdx.x * 256 + threadIdx.x;
    if (i < K * N) {
        int k = i / N, n = i - k * N;
        d[(size_t)n * K + k] = (_Float16)s[i];
    }
}

// ------------------------------------------------------- LayerNorm (+shift)
// One wave per token; SHIFT=true applies cyclic shift + window partition:
// output row m = (b*64 + win)*49 + n reads source token ((rs+3)%56,(cs+3)%56).
template<bool SHIFT>
__global__ __launch_bounds__(256) void ln_kernel(
    const float* __restrict__ x, const float* __restrict__ g,
    const float* __restrict__ b, _Float16* __restrict__ out)
{
    int gw   = (blockIdx.x * 256 + threadIdx.x) >> 5;
    int lane = threadIdx.x & 31;
    size_t dst = (size_t)gw * 384;
    size_t src;
    if (SHIFT) {
        int bi  = gw / 3136;
        int rem = gw - bi * 3136;
        int w   = rem / 49, n = rem - w * 49;
        int rs  = (w >> 3) * 7 + n / 7;
        int cs  = (w & 7) * 7 + n % 7;
        int rr  = rs + 3; if (rr >= 56) rr -= 56;
        int cc  = cs + 3; if (cc >= 56) cc -= 56;
        src = ((size_t)bi * 3136 + rr * 56 + cc) * 384;
    } else {
        src = dst;
    }
    float vals[12];
    float s = 0.f;
#pragma unroll
    for (int e = 0; e < 12; ++e) { vals[e] = x[src + lane + 32 * e]; s += vals[e]; }
#pragma unroll
    for (int m = 16; m >= 1; m >>= 1) s += __shfl_xor(s, m, 32);
    float mean = s * (1.0f / 384.0f);
    float s2 = 0.f;
#pragma unroll
    for (int e = 0; e < 12; ++e) { float d0 = vals[e] - mean; s2 += d0 * d0; }
#pragma unroll
    for (int m = 16; m >= 1; m >>= 1) s2 += __shfl_xor(s2, m, 32);
    float rstd = rsqrtf(s2 * (1.0f / 384.0f) + 1e-5f);
#pragma unroll
    for (int e = 0; e < 12; ++e) {
        int ch = lane + 32 * e;
        out[dst + ch] = (_Float16)((vals[e] - mean) * rstd * g[ch] + b[ch]);
    }
}

// ------------------------------------------------------------- WMMA GEMM
// C = A(f16, MxK, row-major) * Bt(f16, NxK, row-major = B^T) + bias.
// Double-buffered LDS tiles filled by async global->LDS DMA.
// EPI 0: f16 store          EPI 1: exact GELU -> f16 store
// EPI 2: proj (window-reverse + unshift + residual -> f32)
// EPI 3: residual add -> f32 (final output)
template<int EPI>
__global__ __launch_bounds__(256) void gemm_wmma(
    const _Float16* __restrict__ A, const _Float16* __restrict__ Bt,
    const float* __restrict__ bias,
    _Float16* __restrict__ outH, float* __restrict__ outF,
    const float* __restrict__ resid,
    int M, int N, int K)
{
    __shared__ __align__(16) _Float16 As[2][BM][BK + 8];    // 2 x 64 x 40
    __shared__ __align__(16) _Float16 Bs[2][BN][BK + 8];    // 2 x 128 x 40 (col-major tile)

    const int tid  = threadIdx.x;
    const int wave = tid >> 5, lane = tid & 31;
    const int wm = wave >> 2, wn = wave & 3;                // 2 x 4 wave grid
    const int m0 = blockIdx.y * BM;
    const int n0 = blockIdx.x * BN;

    v8f acc[2][2] = {{{}, {}}, {{}, {}}};

    // staging coordinates (16B chunks per lane)
    const int ar = tid >> 2;            // A: row 0..63
    const int ac = (tid & 3) * 8;       // A: k chunk {0,8,16,24}
    const int bcol = tid >> 1;          // B: col 0..127
    const int bk  = (tid & 1) * 16;     // B: k chunk {0,16}

    const int nk = K / BK;

    // stage tile `it` into buffer `buf`
    auto stage = [&](int it, int buf) {
        const int k0 = it * BK;
        async_copy_b128(&As[buf][ar][ac],      A  + (size_t)(m0 + ar)   * K + k0 + ac);
        async_copy_b128(&Bs[buf][bcol][bk],    Bt + (size_t)(n0 + bcol) * K + k0 + bk);
        async_copy_b128(&Bs[buf][bcol][bk + 8], Bt + (size_t)(n0 + bcol) * K + k0 + bk + 8);
    };

    stage(0, 0);

    const int arow0 = wm * 32 + (lane & 15);
    const int kg    = (lane >> 4) * 8;
    const int bcol0 = wn * 32 + (lane & 15);
    const int kb    = (lane >> 4) * 8;

    for (int it = 0; it < nk; ++it) {
        const int buf = it & 1;
        wait_async0();              // my async copies for this tile have landed
        __syncthreads();            // everyone's have landed; prev compute done
        if (it + 1 < nk) stage(it + 1, buf ^ 1);   // overlap DMA with WMMAs

        HF16 afr[2], bfr[2];
#pragma unroll
        for (int mt = 0; mt < 2; ++mt) {
            afr[mt].h2[0] = *(const h8*)&As[buf][arow0 + mt * 16][kg];
            afr[mt].h2[1] = *(const h8*)&As[buf][arow0 + mt * 16][16 + kg];
        }
#pragma unroll
        for (int nt = 0; nt < 2; ++nt) {
            bfr[nt].h2[0] = *(const h8*)&Bs[buf][bcol0 + nt * 16][kb];
            bfr[nt].h2[1] = *(const h8*)&Bs[buf][bcol0 + nt * 16][16 + kb];
        }
#pragma unroll
        for (int mt = 0; mt < 2; ++mt)
#pragma unroll
            for (int nt = 0; nt < 2; ++nt)
                acc[mt][nt] = __builtin_amdgcn_wmma_f32_16x16x32_f16(
                    false, afr[mt].v, false, bfr[nt].v,
                    (short)0, acc[mt][nt], false, false);
    }

#pragma unroll
    for (int mt = 0; mt < 2; ++mt)
#pragma unroll
        for (int nt = 0; nt < 2; ++nt)
#pragma unroll
            for (int r = 0; r < 8; ++r) {
                int row = m0 + wm * 32 + mt * 16 + r + ((lane >> 4) * 8);
                int col = n0 + wn * 32 + nt * 16 + (lane & 15);
                float v = acc[mt][nt][r] + bias[col];
                if (EPI == 0) {
                    outH[(size_t)row * N + col] = (_Float16)v;
                } else if (EPI == 1) {
                    float gl = 0.5f * v * (1.0f + erff(v * 0.70710678118654752f));
                    outH[(size_t)row * N + col] = (_Float16)gl;
                } else if (EPI == 2) {
                    int b   = row / 3136;
                    int rem = row - b * 3136;
                    int w   = rem / 49;
                    int n   = rem - w * 49;
                    int rs  = (w >> 3) * 7 + n / 7;
                    int cs  = (w & 7) * 7 + n % 7;
                    int rr  = rs + 3; if (rr >= 56) rr -= 56;
                    int cc  = cs + 3; if (cc >= 56) cc -= 56;
                    size_t idx = ((size_t)b * 3136 + rr * 56 + cc) * 384 + col;
                    outF[idx] = resid[idx] + v;
                } else {
                    size_t idx = (size_t)row * N + col;
                    outF[idx] = resid[idx] + v;
                }
            }
}

// --------------------------------------------------------------- attention
__device__ __forceinline__ int regid(int g) {
    return (g < 49) ? 0 : ((g < 53) ? 1 : 2);
}

// One block (4 waves) per (window, head). 49x49 attention padded to 64x64.
__global__ __launch_bounds__(128) void attn_kernel(
    const _Float16* __restrict__ qkv,        // [100352, 1152]
    const float* __restrict__ bias_table,    // [169, 12]
    _Float16* __restrict__ out)              // [100352, 384]
{
    __shared__ __align__(16) _Float16 Qs[64][40];
    __shared__ __align__(16) _Float16 Ks[64][40];
    __shared__ __align__(16) _Float16 VsT[32][72];   // v transposed [d][n]
    __shared__ float Sf[64][65];
    __shared__ __align__(16) _Float16 Ps[64][72];

    const int wfull = blockIdx.x / 12;               // window 0..2047
    const int h     = blockIdx.x % 12;
    const int tid   = threadIdx.x;
    const int wave  = tid >> 5, lane = tid & 31;

    // ---- stage Q, K (row-major, zero-padded) and V (transposed)
    {
        h8 z = {};
        int r = tid >> 1, hf = tid & 1, c0 = hf * 16;
        h8 q0 = z, q1 = z, k0 = z, k1 = z;
        if (r < 49) {
            const _Float16* base = qkv + (size_t)(wfull * 49 + r) * 1152 + h * 32;
            q0 = *(const h8*)(base + c0);
            q1 = *(const h8*)(base + c0 + 8);
            k0 = *(const h8*)(base + 384 + c0);
            k1 = *(const h8*)(base + 384 + c0 + 8);
        }
        *(h8*)&Qs[r][c0] = q0;  *(h8*)&Qs[r][c0 + 8] = q1;
        *(h8*)&Ks[r][c0] = k0;  *(h8*)&Ks[r][c0 + 8] = k1;

        int vr = tid & 63, vh = tid >> 6, vc0 = vh * 16;
        h8 v0 = z, v1 = z;
        if (vr < 49) {
            const _Float16* vb = qkv + (size_t)(wfull * 49 + vr) * 1152 + h * 32 + 768;
            v0 = *(const h8*)(vb + vc0);
            v1 = *(const h8*)(vb + vc0 + 8);
        }
#pragma unroll
        for (int e = 0; e < 8; ++e) { VsT[vc0 + e][vr] = v0[e]; VsT[vc0 + 8 + e][vr] = v1[e]; }
    }
    __syncthreads();

    // ---- scores = Q K^T * scale + rel-bias + sw-mask
    {
        const int rbase = wave * 16;
        const int arow  = rbase + (lane & 15);
        const int kg    = (lane >> 4) * 8;
        HF16 afr;
        afr.h2[0] = *(const h8*)&Qs[arow][kg];
        afr.h2[1] = *(const h8*)&Qs[arow][16 + kg];
        const int kb  = (lane >> 4) * 8;
        const int win = wfull & 63;
        const int wh  = win >> 3, ww = win & 7;
#pragma unroll
        for (int nt = 0; nt < 4; ++nt) {
            HF16 bfr;
            int col = nt * 16 + (lane & 15);
            bfr.h2[0] = *(const h8*)&Ks[col][kb];
            bfr.h2[1] = *(const h8*)&Ks[col][16 + kb];
            v8f acc = {};
            acc = __builtin_amdgcn_wmma_f32_16x16x32_f16(
                false, afr.v, false, bfr.v, (short)0, acc, false, false);
#pragma unroll
            for (int r = 0; r < 8; ++r) {
                int row = rbase + r + ((lane >> 4) * 8);
                float s;
                if (row < 49 && col < 49) {
                    int ri = row / 7, rj = row % 7;
                    int ci = col / 7, cj = col % 7;
                    int d0 = ri - ci + 6, d1 = rj - cj + 6;
                    float bi = bias_table[(d0 * 13 + d1) * 12 + h];
                    int cr = 3 * regid(wh * 7 + ri) + regid(ww * 7 + rj);
                    int cn = 3 * regid(wh * 7 + ci) + regid(ww * 7 + cj);
                    float mk = (cr != cn) ? -100.0f : 0.0f;
                    s = acc[r] * 0.17677669529663687f + bi + mk;
                } else {
                    s = -1e30f;
                }
                Sf[row][col] = s;
            }
        }
    }
    __syncthreads();

    // ---- softmax over valid 49 columns; probs -> f16 (padded with zeros)
    if (tid < 64) {
        int r = tid;
        if (r < 49) {
            float mx = -1e30f;
            for (int c = 0; c < 49; ++c) mx = fmaxf(mx, Sf[r][c]);
            float sum = 0.f;
            for (int c = 0; c < 49; ++c) {
                float e = __expf(Sf[r][c] - mx);
                Sf[r][c] = e;
                sum += e;
            }
            float inv = 1.0f / sum;
            for (int c = 0; c < 64; ++c)
                Ps[r][c] = (c < 49) ? (_Float16)(Sf[r][c] * inv) : (_Float16)0.f;
        } else {
            for (int c = 0; c < 64; ++c) Ps[r][c] = (_Float16)0.f;
        }
    }
    __syncthreads();

    // ---- out = P @ V   (K=64 -> two k-steps of 32, N=32 -> two tiles)
    {
        const int rbase = wave * 16;
        const int arow  = rbase + (lane & 15);
        const int kg8   = (lane >> 4) * 8;
        const int kg16  = (lane >> 4) * 16;
        v8f oacc[2] = {{}, {}};
#pragma unroll
        for (int ks = 0; ks < 2; ++ks) {
            HF16 afr;
            afr.h2[0] = *(const h8*)&Ps[arow][ks * 32 + kg8];
            afr.h2[1] = *(const h8*)&Ps[arow][ks * 32 + 16 + kg8];
#pragma unroll
            for (int nt = 0; nt < 2; ++nt) {
                HF16 bfr;
                int col = nt * 16 + (lane & 15);
                bfr.h2[0] = *(const h8*)&VsT[col][ks * 32 + kg8];
                bfr.h2[1] = *(const h8*)&VsT[col][ks * 32 + 16 + kg8];
                oacc[nt] = __builtin_amdgcn_wmma_f32_16x16x32_f16(
                    false, afr.v, false, bfr.v, (short)0, oacc[nt], false, false);
            }
        }
#pragma unroll
        for (int nt = 0; nt < 2; ++nt)
#pragma unroll
            for (int r = 0; r < 8; ++r) {
                int row = rbase + r + ((lane >> 4) * 8);
                if (row < 49) {
                    int col = nt * 16 + (lane & 15);
                    out[(size_t)(wfull * 49 + row) * 384 + h * 32 + col] = (_Float16)oacc[nt][r];
                }
            }
    }
}

// ---------------------------------------------------------------- launcher
extern "C" void kernel_launch(void* const* d_in, const int* in_sizes, int n_in,
                              void* d_out, int out_size, void* d_ws, size_t ws_size,
                              hipStream_t stream) {
    const float* x          = (const float*)d_in[0];
    const float* W_qkv      = (const float*)d_in[1];
    const float* b_qkv      = (const float*)d_in[2];
    const float* W_proj     = (const float*)d_in[3];
    const float* b_proj     = (const float*)d_in[4];
    const float* bias_table = (const float*)d_in[5];
    const float* gamma1     = (const float*)d_in[6];
    const float* beta1      = (const float*)d_in[7];
    const float* gamma2     = (const float*)d_in[8];
    const float* beta2      = (const float*)d_in[9];
    const float* W_mlp1     = (const float*)d_in[10];
    const float* b_mlp1     = (const float*)d_in[11];
    const float* W_mlp2     = (const float*)d_in[12];
    const float* b_mlp2     = (const float*)d_in[13];

    const int M = 100352;           // 32 * 56 * 56

    char* ws = (char*)d_ws;
    // region A: [0, 308281344) = XW (77MB) + QKV (231MB); reused by HID (308MB)
    _Float16* XW   = (_Float16*)(ws + 0);
    _Float16* QKV  = (_Float16*)(ws + 77070336ull);
    _Float16* HID  = (_Float16*)(ws + 0);
    // region B: attn output; reused by H2
    _Float16* ATTN = (_Float16*)(ws + 308281344ull);
    _Float16* H2   = ATTN;
    // region C: f32 residual x1
    float* X1 = (float*)(ws + 385351680ull);
    // region D: f16 transposed weights [N,K]
    _Float16* Wq16  = (_Float16*)(ws + 539492352ull);
    _Float16* Wp16  = Wq16  + 1152 * 384;
    _Float16* Wm116 = Wp16  + 384 * 384;
    _Float16* Wm216 = Wm116 + 1536 * 384;

    // weight conversion + transpose: [K,N] f32 -> [N,K] f16
    transpose_f32_to_f16<<<(384 * 1152 + 255) / 256, 256, 0, stream>>>(W_qkv,  Wq16,  384, 1152);
    transpose_f32_to_f16<<<(384 * 384  + 255) / 256, 256, 0, stream>>>(W_proj, Wp16,  384, 384);
    transpose_f32_to_f16<<<(384 * 1536 + 255) / 256, 256, 0, stream>>>(W_mlp1, Wm116, 384, 1536);
    transpose_f32_to_f16<<<(1536 * 384 + 255) / 256, 256, 0, stream>>>(W_mlp2, Wm216, 1536, 384);

    // LN1 + shift + window partition
    ln_kernel<true><<<M / 8, 256, 0, stream>>>(x, gamma1, beta1, XW);

    // QKV GEMM: [M,384] x [384,1152]
    gemm_wmma<0><<<dim3(1152 / BN, M / BM), 256, 0, stream>>>(
        XW, Wq16, b_qkv, QKV, nullptr, nullptr, M, 1152, 384);

    // attention: one block per (window, head)
    attn_kernel<<<2048 * 12, 128, 0, stream>>>(QKV, bias_table, ATTN);

    // proj GEMM + window-reverse/unshift + residual -> X1 (f32)
    gemm_wmma<2><<<dim3(384 / BN, M / BM), 256, 0, stream>>>(
        ATTN, Wp16, b_proj, nullptr, X1, x, M, 384, 384);

    // LN2
    ln_kernel<false><<<M / 8, 256, 0, stream>>>(X1, gamma2, beta2, H2);

    // MLP1 GEMM + GELU
    gemm_wmma<1><<<dim3(1536 / BN, M / BM), 256, 0, stream>>>(
        H2, Wm116, b_mlp1, HID, nullptr, nullptr, M, 1536, 384);

    // MLP2 GEMM + residual -> d_out (f32)
    gemm_wmma<3><<<dim3(384 / BN, M / BM), 256, 0, stream>>>(
        HID, Wm216, b_mlp2, nullptr, (float*)d_out, X1, M, 384, 1536);
}
